// Conv2DMod_4879082849036
// MI455X (gfx1250) — compile-verified, matrix operand order checked by reading, not run on hardware
//
#include <hip/hip_runtime.h>

// ---------------------------------------------------------------------------
// Conv2DMod (StyleGAN modulated conv) for gfx1250 via implicit-GEMM WMMA.
// B=16, H=W=128, Cin=F=256, K=3.  GEMM: M=B*H*W, N=256, Kdim=2304.
// bf16 A/B fragments, f32 accumulation with v_wmma_f32_16x16x32_bf16.
// Weights pre-converted to bf16 [kidx][f][c] in d_ws by the prep kernel;
// B staging uses async global->LDS copies (ASYNCcnt) when available.
// ---------------------------------------------------------------------------

typedef __attribute__((ext_vector_type(16))) __bf16 v16bf;
typedef __attribute__((ext_vector_type(8)))  float  v8f;
typedef int v4ib __attribute__((vector_size(16)));   // matches builtin pointee

#define EPS_     1e-8f
#define CIN_     256
#define FOUT_    256
#define H_       128
#define W_       128
#define B_       16
#define AROW     36   // A LDS row: 32 bf16 + 4 pad = 72 B (conflict-free frag reads)
#define BROW     40   // B LDS row: 32 bf16 + 8 pad = 80 B (16B-aligned rows)
#define WS_BW_OFF 2048  // byte offset of bf16 weights inside d_ws

#if defined(__has_builtin)
#if __has_builtin(__builtin_amdgcn_global_load_async_to_lds_b128)
#define HAS_ASYNC_LDS 1
#endif
#endif

// global (as1) / LDS (as3) pointers to 4xint vectors, per builtin signature
#define GAS1(p) ((__attribute__((address_space(1))) v4ib*)(void*)(p))
#define LAS3(p) ((__attribute__((address_space(3))) v4ib*)(void*)(p))

__device__ __forceinline__ unsigned int pack2bf(float a, float b) {
    unsigned short ua = __builtin_bit_cast(unsigned short, (__bf16)a);
    unsigned short ub = __builtin_bit_cast(unsigned short, (__bf16)b);
    return (unsigned int)ua | ((unsigned int)ub << 16);
}

// ---------------------------------------------------------------------------
// Prep: block 0   -> stats[0..255]=||W_f||^2, stats[256..271]=||s_b||^2
//       blocks 1..9 -> bf16 weights transposed to [kidx][f][c] at ws+2048
// ---------------------------------------------------------------------------
__global__ void conv2dmod_prep(const float* __restrict__ kern,
                               const float* __restrict__ style,
                               float* __restrict__ stats,
                               __bf16* __restrict__ bw) {
    const int t = threadIdx.x;  // 256 threads
    if (blockIdx.x == 0) {
        float s = 0.0f;
        for (int i = 0; i < 9 * CIN_; ++i) {
            float v = kern[(size_t)i * FOUT_ + t];
            s += v * v;
        }
        stats[t] = s;
        if (t < B_) {
            float ss = 0.0f;
            for (int c = 0; c < CIN_; ++c) {
                float v = style[t * CIN_ + c];
                ss += v * v;
            }
            stats[FOUT_ + t] = ss;
        }
    } else {
        const int kidx = blockIdx.x - 1;                 // 0..8
        const float* src = kern + (size_t)kidx * CIN_ * FOUT_;
        __bf16* drow = bw + ((size_t)kidx * FOUT_ + t) * CIN_;  // row f=t
        for (int c = 0; c < CIN_; ++c)
            drow[c] = (__bf16)src[(size_t)c * FOUT_ + t];
    }
}

// ---------------------------------------------------------------------------
// Main kernel: one block per (b, h) output row.
//   M-tile = 128 (all w), N-tile = 256 (all f), K chunks of 32 channels.
//   8 waves arranged 2(M) x 4(N); each wave computes 64x64 via 16 WMMA accs.
// ---------------------------------------------------------------------------
__global__ void __launch_bounds__(256)
conv2dmod_wmma(const float* __restrict__ x,
               const float* __restrict__ style,
               const __bf16* __restrict__ bw,
               const float* __restrict__ stats,
               float* __restrict__ out) {
    __shared__ float  s_style[CIN_];
    __shared__ __align__(16) __bf16 At[128 * AROW];   //  9216 B
    __shared__ __align__(16) __bf16 Bt[256 * BROW];   // 20480 B

    const int tid = threadIdx.x;
    const int bh  = blockIdx.x;
    const int b   = bh >> 7;       // / 128
    const int h   = bh & 127;

    s_style[tid] = 1.0f + style[b * CIN_ + tid];

    const int lane  = tid & 31;
    const int wid   = tid >> 5;
    const int mBase = (wid >> 2) * 64;   // 0 or 64
    const int nBase = (wid & 3) * 64;    // 0,64,128,192
    const int mrow  = lane & 15;         // row/col within a 16x16 frag
    const int sel   = lane >> 4;         // K-half select per ISA A/B layout

    const int p    = tid >> 1;           // pixel 0..127 for A staging
    const int half = tid & 1;            // which 16-channel half

    v8f acc[4][4];
    #pragma unroll
    for (int i = 0; i < 4; ++i)
        #pragma unroll
        for (int j = 0; j < 4; ++j)
            acc[i][j] = (v8f)(0.0f);

    __syncthreads();  // s_style visible

    for (int kh = 0; kh < 3; ++kh) {
        int hh = h + kh - 1;                       // symmetric pad p=1 == clamp
        hh = hh < 0 ? 0 : (hh > H_ - 1 ? H_ - 1 : hh);
        const float* xrow = x + (((size_t)b * H_ + hh) * W_) * CIN_;

        for (int kw = 0; kw < 3; ++kw) {
            const int kidx = kh * 3 + kw;
            int wc = p + kw - 1;
            wc = wc < 0 ? 0 : (wc > W_ - 1 ? W_ - 1 : wc);
            const float* xpix = xrow + (size_t)wc * CIN_;
            // bf16 weights, row f = tid, layout [kidx][f][c]
            const __bf16* brow = bw + ((size_t)kidx * FOUT_ + tid) * CIN_;

            for (int c0 = 0; c0 < CIN_; c0 += 32) {
                // ---- stage A: modulated x -> bf16 LDS [p][k] ----
                {
                    const float4* s4 = (const float4*)(xpix + c0 + half * 16);
                    float4 v[4];
                    #pragma unroll
                    for (int q = 0; q < 4; ++q) v[q] = s4[q];
                    const float* st = &s_style[c0 + half * 16];
                    unsigned long long* dst =
                        (unsigned long long*)&At[p * AROW + half * 16];
                    #pragma unroll
                    for (int q = 0; q < 4; ++q) {
                        float f0 = v[q].x * st[4 * q + 0];
                        float f1 = v[q].y * st[4 * q + 1];
                        float f2 = v[q].z * st[4 * q + 2];
                        float f3 = v[q].w * st[4 * q + 3];
                        unsigned long long qq =
                            (unsigned long long)pack2bf(f0, f1) |
                            ((unsigned long long)pack2bf(f2, f3) << 32);
                        dst[q] = qq;
                    }
                }
                // ---- stage B: bf16 weights -> LDS [f][k], pure 64B copy ----
                {
                    const __bf16* bsrc = brow + c0;        // 64 B, 16B-aligned
                    __bf16*       bdst = &Bt[tid * BROW];  // 80B rows, 16B-aligned
#if defined(HAS_ASYNC_LDS)
                    // IOFFSET is added to both global and LDS addresses.
                    __builtin_amdgcn_global_load_async_to_lds_b128(
                        GAS1(bsrc), LAS3(bdst), 0, 0);
                    __builtin_amdgcn_global_load_async_to_lds_b128(
                        GAS1(bsrc), LAS3(bdst), 16, 0);
                    __builtin_amdgcn_global_load_async_to_lds_b128(
                        GAS1(bsrc), LAS3(bdst), 32, 0);
                    __builtin_amdgcn_global_load_async_to_lds_b128(
                        GAS1(bsrc), LAS3(bdst), 48, 0);
#else
                    const unsigned long long* g8 = (const unsigned long long*)bsrc;
                    unsigned long long*       d8 = (unsigned long long*)bdst;
                    #pragma unroll
                    for (int q = 0; q < 8; ++q) d8[q] = g8[q];
#endif
                }
#if defined(HAS_ASYNC_LDS)
#if __has_builtin(__builtin_amdgcn_s_wait_asynccnt)
                __builtin_amdgcn_s_wait_asynccnt(0);
#else
                asm volatile("s_wait_asynccnt 0x0" ::: "memory");
#endif
#endif
                __syncthreads();

                // ---- gather fragments per ISA 16-bit 16x32 layout ----
                v16bf afr[4], bfr[4];
                #pragma unroll
                for (int im = 0; im < 4; ++im) {
                    const unsigned long long* q0 = (const unsigned long long*)
                        &At[(mBase + im * 16 + mrow) * AROW + sel * 8];
                    union { v16bf v; unsigned long long q[4]; } u;
                    u.q[0] = q0[0]; u.q[1] = q0[1];   // K 0..7  (or 8..15)
                    u.q[2] = q0[4]; u.q[3] = q0[5];   // K 16..23 (or 24..31)
                    afr[im] = u.v;
                }
                #pragma unroll
                for (int in = 0; in < 4; ++in) {
                    const unsigned long long* q0 = (const unsigned long long*)
                        &Bt[(nBase + in * 16 + mrow) * BROW + sel * 8];
                    union { v16bf v; unsigned long long q[4]; } u;
                    u.q[0] = q0[0]; u.q[1] = q0[1];
                    u.q[2] = q0[4]; u.q[3] = q0[5];
                    bfr[in] = u.v;
                }

                // ---- 16 WMMAs: 64x64 wave tile, K += 32 ----
                #pragma unroll
                for (int im = 0; im < 4; ++im)
                    #pragma unroll
                    for (int in = 0; in < 4; ++in)
                        acc[im][in] = __builtin_amdgcn_wmma_f32_16x16x32_bf16(
                            false, afr[im], false, bfr[in],
                            (short)0, acc[im][in], false, false);

                __syncthreads();  // before next staging overwrites tiles
            }
        }
    }

    // ---- epilogue: demodulate and store f32 ----
    const float ssq = stats[FOUT_ + b];
    float scale[4];
    #pragma unroll
    for (int in = 0; in < 4; ++in) {
        int n = nBase + in * 16 + mrow;
        scale[in] = rsqrtf(stats[n] * (float)(H_ * W_) + ssq + EPS_);
    }
    #pragma unroll
    for (int im = 0; im < 4; ++im) {
        #pragma unroll
        for (int r = 0; r < 8; ++r) {
            // C/D layout: VGPR r -> M = r (lanes 0-15) or r+8 (lanes 16-31)
            int m = mBase + im * 16 + r + 8 * sel;
            float* orow = out + (((size_t)b * H_ + h) * W_ + m) * FOUT_;
            #pragma unroll
            for (int in = 0; in < 4; ++in) {
                int n = nBase + in * 16 + mrow;
                orow[n] = acc[im][in][r] * scale[in];
            }
        }
    }
}

// ---------------------------------------------------------------------------
extern "C" void kernel_launch(void* const* d_in, const int* in_sizes, int n_in,
                              void* d_out, int out_size, void* d_ws, size_t ws_size,
                              hipStream_t stream) {
    const float* x     = (const float*)d_in[0];  // [16,128,128,256]
    const float* style = (const float*)d_in[1];  // [16,1,1,256]
    const float* kern  = (const float*)d_in[2];  // [3,3,256,256] HWIO
    float*  out   = (float*)d_out;               // [16,128,128,256]
    float*  stats = (float*)d_ws;                // 256 w_sq + 16 s_sq floats
    __bf16* bwgt  = (__bf16*)((char*)d_ws + WS_BW_OFF);  // 9*256*256 bf16

    conv2dmod_prep<<<10, 256, 0, stream>>>(kern, style, stats, bwgt);
    conv2dmod_wmma<<<B_ * H_, 256, 0, stream>>>(x, style, bwgt, stats, out);
}
